// MaskedMultiHeadAttention_27436251087289
// MI455X (gfx1250) — matrix operand order repaired and checked by reading, not compiled, not run on hardware
//
#include <hip/hip_runtime.h>

#define B_  4
#define L_  2048
#define D_  512
#define H_  8
#define DKK 64

typedef __bf16 bf16_t;
typedef __attribute__((ext_vector_type(16))) __bf16 v16bf;
typedef __attribute__((ext_vector_type(8)))  __bf16 v8bf;
typedef __attribute__((ext_vector_type(8)))  float  v8f;
typedef int v4i_vs __attribute__((vector_size(16)));   // matches builtin param type

#define AS1 __attribute__((address_space(1)))
#define AS3 __attribute__((address_space(3)))

static __device__ __forceinline__ unsigned short f2bfraw(float f) {
    unsigned u = __builtin_bit_cast(unsigned, f);
    u += 0x7FFFu + ((u >> 16) & 1u);          // round-to-nearest-even
    return (unsigned short)(u >> 16);
}
static __device__ __forceinline__ __bf16 f2bf(float f) {
    unsigned short s = f2bfraw(f);
    return __builtin_bit_cast(__bf16, s);
}
static __device__ __forceinline__ v16bf cat16(v8bf lo, v8bf hi) {
    return __builtin_shufflevector(lo, hi, 0,1,2,3,4,5,6,7,8,9,10,11,12,13,14,15);
}
static __device__ __forceinline__ v8f vzero8() {
    v8f z;
#pragma unroll
    for (int i = 0; i < 8; ++i) z[i] = 0.0f;
    return z;
}
static __device__ __forceinline__ v8f wmma_bf16(v16bf a, v16bf b, v8f c) {
    return __builtin_amdgcn_wmma_f32_16x16x32_bf16(false, a, false, b, (short)0, c, false, false);
}

// 16-byte global -> LDS copy: async on gfx1250 toolchains that expose the
// builtin (ASYNCcnt-tracked global_load_async_to_lds_b128), else synchronous.
static __device__ __forceinline__ void async_cp16(const bf16_t* g, void* l) {
#if __has_builtin(__builtin_amdgcn_global_load_async_to_lds_b128)
    __builtin_amdgcn_global_load_async_to_lds_b128(
        (AS1 v4i_vs*)(unsigned long long)g,
        (AS3 v4i_vs*)(unsigned)(unsigned long long)l,
        0, 0);
#else
    *(v8bf*)l = *(const v8bf*)g;
#endif
}
static __device__ __forceinline__ void wait_async0() {
#if __has_builtin(__builtin_amdgcn_s_wait_asynccnt)
    __builtin_amdgcn_s_wait_asynccnt(0);
#else
    asm volatile("s_wait_asynccnt 0x0" ::: "memory");
#endif
}

// ---------------------------------------------------------------------------
// Kernel 0: x,y f32 -> bf16 once (kills redundant per-wave VALU conversion
// and halves GEMM A-operand bytes).
// ---------------------------------------------------------------------------
__global__ __launch_bounds__(256) void cvt_xy(
    const float* __restrict__ x, const float* __restrict__ y,
    bf16_t* __restrict__ xb, bf16_t* __restrict__ yb)
{
    const size_t i = ((size_t)blockIdx.x * 256 + threadIdx.x) * 8;
    const float* s = blockIdx.y ? y : x;
    bf16_t*      d = blockIdx.y ? yb : xb;
    v8bf o;
#pragma unroll
    for (int j = 0; j < 8; ++j) o[j] = f2bf(s[i + j]);
    *(v8bf*)(d + i) = o;
}

// ---------------------------------------------------------------------------
// Kernel 1: fold mask into W, transpose, cast to bf16.
// ---------------------------------------------------------------------------
__global__ __launch_bounds__(256) void prep_weights(
    const float* __restrict__ Wq, const float* __restrict__ Wk,
    const float* __restrict__ Wv, const float* __restrict__ Wo,
    const float* __restrict__ mask, bf16_t* __restrict__ wt)
{
    const int idx = blockIdx.x * 256 + threadIdx.x;       // 0 .. D*D-1
    const int mat = blockIdx.y;                           // 0..3
    const float* W = (mat == 0) ? Wq : (mat == 1) ? Wk : (mat == 2) ? Wv : Wo;
    const int o = idx % D_;
    const int i = idx / D_;
    const float v = W[(size_t)i * D_ + o] * mask[(size_t)i * D_ + o];
    wt[(size_t)mat * D_ * D_ + (size_t)o * D_ + i] = f2bf(v);
}

// ---------------------------------------------------------------------------
// Kernel 2: Q/K/V projections (bf16 WMMA, f32 accumulate), A already bf16.
//   mode 0: Q (scaled by sqrt(DK)=8)  -> [B,H,L,DK]
//   mode 1: K                          -> [B,H,L,DK]
//   mode 2: V                          -> transposed [B,H,DK,L]
// ---------------------------------------------------------------------------
__global__ __launch_bounds__(256) void proj_qkv(
    const bf16_t* __restrict__ xb, const bf16_t* __restrict__ yb,
    const bf16_t* __restrict__ wt, bf16_t* __restrict__ qkv)
{
    const int mode = blockIdx.y;
    const bf16_t* src = (mode == 0) ? xb : yb;
    const bf16_t* WT  = wt + (size_t)mode * D_ * D_;
    bf16_t*       dst = qkv + (size_t)mode * ((size_t)B_ * H_ * L_ * DKK);

    const int lane  = threadIdx.x & 31;
    const int wave  = threadIdx.x >> 5;
    const int wid   = blockIdx.x * 8 + wave;
    const int ngrp  = wid & 7;
    const int mtile = wid >> 3;
    const int tok0  = mtile * 16;
    const int col16 = lane & 15;
    const int half  = lane >> 4;

    v8f acc[4];
#pragma unroll
    for (int nt = 0; nt < 4; ++nt) acc[nt] = vzero8();

    const bf16_t* arow = src + (size_t)(tok0 + col16) * D_;

    for (int kc = 0; kc < D_ / 32; ++kc) {
        v8bf lo = *(const v8bf*)(arow + kc * 32 + half * 16);
        v8bf hi = *(const v8bf*)(arow + kc * 32 + half * 16 + 8);
        v16bf a = cat16(lo, hi);
#pragma unroll
        for (int nt = 0; nt < 4; ++nt) {
            const bf16_t* bp = WT + (size_t)(ngrp * 64 + nt * 16 + col16) * D_
                                  + kc * 32 + half * 16;
            v16bf bm = *(const v16bf*)bp;
            acc[nt] = wmma_bf16(a, bm, acc[nt]);
        }
    }

    const float sc = (mode == 0) ? 8.0f : 1.0f;   // fold quirky *sqrt(dk)
    const int h = ngrp;
#pragma unroll
    for (int nt = 0; nt < 4; ++nt) {
#pragma unroll
        for (int r = 0; r < 8; ++r) {
            const int tok = tok0 + r + half * 8;
            const int bi = tok / L_, li = tok % L_;
            const int dk = nt * 16 + col16;
            const float v = acc[nt][r] * sc;
            size_t addr;
            if (mode < 2)
                addr = (((size_t)(bi * H_ + h)) * L_ + li) * DKK + dk;   // [B,H,L,DK]
            else
                addr = (((size_t)(bi * H_ + h)) * DKK + dk) * L_ + li;   // [B,H,DK,L]
            dst[addr] = f2bf(v);
        }
    }
}

// ---------------------------------------------------------------------------
// Kernel 3: flash attention with async double-buffered K/V staging in LDS.
// All 8 waves of a block share (b,h): one cooperative 8KB async tile fetch per
// 32-key step feeds every wave's WMMA fragments (8x less L2 traffic).
// ---------------------------------------------------------------------------
__global__ __launch_bounds__(256) void flash_attn(
    const bf16_t* __restrict__ Qw, const bf16_t* __restrict__ Kw,
    const bf16_t* __restrict__ VTw, bf16_t* __restrict__ Zw)
{
    __shared__ alignas(16) unsigned short Kt[2][32][64];     // [buf][key][dk]
    __shared__ alignas(16) unsigned short Vt[2][64][32];     // [buf][dk][key]
    __shared__ alignas(16) unsigned short plds[8][16][32];   // per-wave P tile

    const int tid   = threadIdx.x;
    const int lane  = tid & 31;
    const int wave  = tid >> 5;
    const int col16 = lane & 15;
    const int half  = lane >> 4;

    const int nqblk = L_ / 128;
    const int bh = blockIdx.x / nqblk;
    const int q0 = (blockIdx.x % nqblk) * 128 + wave * 16;

    const bf16_t* Qb = Qw  + (size_t)bh * L_ * DKK;
    const bf16_t* Kb = Kw  + (size_t)bh * L_ * DKK;
    const bf16_t* Vb = VTw + (size_t)bh * DKK * L_;

    // cooperative tile loader: 256 threads x 2 b128 copies = 8KB per tile
    const int krow = tid >> 3, kch = tid & 7;    // K tile 32x64 bf16
    const int vrow = tid >> 2, vch = tid & 3;    // V tile 64x32 bf16
    auto issue_tile = [&](int buf, int kb) {
        async_cp16(Kb + (size_t)(kb + krow) * DKK + kch * 8, &Kt[buf][krow][kch * 8]);
        async_cp16(Vb + (size_t)vrow * L_ + kb + vch * 8,    &Vt[buf][vrow][vch * 8]);
    };

    // Q A-tiles held in registers for the whole key loop (dk 0..63).
    v16bf qa[2];
#pragma unroll
    for (int kc = 0; kc < 2; ++kc) {
        const bf16_t* ap = Qb + (size_t)(q0 + col16) * DKK + kc * 32 + half * 16;
        v8bf lo = *(const v8bf*)ap;
        v8bf hi = *(const v8bf*)(ap + 8);
        qa[kc] = cat16(lo, hi);
    }

    v8f acc[4];
#pragma unroll
    for (int dt = 0; dt < 4; ++dt) acc[dt] = vzero8();
    float m[8], lsum[8];
#pragma unroll
    for (int r = 0; r < 8; ++r) { m[r] = -1e30f; lsum[r] = 0.0f; }

    issue_tile(0, 0);

    for (int it = 0; it < L_ / 32; ++it) {
        const int buf = it & 1;

        wait_async0();                                   // RAW: tile `it` landed
        asm volatile("s_wait_dscnt 0x0" ::: "memory");   // WAR: my LDS reads retired
        __syncthreads();
        if (it + 1 < L_ / 32) issue_tile(buf ^ 1, (it + 1) * 32);  // prefetch

        // ---- S = Q K^T from LDS K tile ----
        v8f S[2];
#pragma unroll
        for (int nt = 0; nt < 2; ++nt) {
            v8f c = vzero8();
#pragma unroll
            for (int kc = 0; kc < 2; ++kc) {
                v16bf bm = *(const v16bf*)&Kt[buf][nt * 16 + col16][kc * 32 + half * 16];
                c = wmma_bf16(qa[kc], bm, c);
            }
            S[nt] = c;
        }

        // ---- online softmax ----
        v8f t;
#pragma unroll
        for (int r = 0; r < 8; ++r) t[r] = fmaxf(S[0][r], S[1][r]);
#pragma unroll
        for (int off = 1; off < 16; off <<= 1)
#pragma unroll
            for (int r = 0; r < 8; ++r)
                t[r] = fmaxf(t[r], __shfl_xor(t[r], off, 32));

        float f[8];
        v8f P0, P1, rs;
#pragma unroll
        for (int r = 0; r < 8; ++r) {
            const float mn = fmaxf(m[r], t[r]);
            f[r]  = __expf(m[r] - mn);
            m[r]  = mn;
            P0[r] = __expf(S[0][r] - mn);
            P1[r] = __expf(S[1][r] - mn);
            rs[r] = P0[r] + P1[r];
        }
#pragma unroll
        for (int off = 1; off < 16; off <<= 1)
#pragma unroll
            for (int r = 0; r < 8; ++r)
                rs[r] += __shfl_xor(rs[r], off, 32);
#pragma unroll
        for (int r = 0; r < 8; ++r) lsum[r] = lsum[r] * f[r] + rs[r];
#pragma unroll
        for (int dt = 0; dt < 4; ++dt)
#pragma unroll
            for (int r = 0; r < 8; ++r) acc[dt][r] *= f[r];

        // ---- P: C-layout -> LDS -> A-layout bf16 (wave-private buffer) ----
        asm volatile("" ::: "memory");
#pragma unroll
        for (int r = 0; r < 8; ++r) {
            plds[wave][r + half * 8][col16]      = f2bfraw(P0[r]);
            plds[wave][r + half * 8][16 + col16] = f2bfraw(P1[r]);
        }
        asm volatile("s_wait_dscnt 0x0" ::: "memory");
        v8bf plo = *(const v8bf*)&plds[wave][col16][half * 16];
        v8bf phi = *(const v8bf*)&plds[wave][col16][half * 16 + 8];
        v16bf pa = cat16(plo, phi);

        // ---- acc += P @ V from LDS V tile ----
#pragma unroll
        for (int dt = 0; dt < 4; ++dt) {
            v16bf bm = *(const v16bf*)&Vt[buf][dt * 16 + col16][half * 16];
            acc[dt] = wmma_bf16(pa, bm, acc[dt]);
        }
    }

    // ---- epilogue: z = acc / lsum -> merged-head bf16 [B,L,D] ----
    float inv[8];
#pragma unroll
    for (int r = 0; r < 8; ++r) inv[r] = 1.0f / lsum[r];
    const int bi = bh / H_, hh = bh % H_;
#pragma unroll
    for (int dt = 0; dt < 4; ++dt)
#pragma unroll
        for (int r = 0; r < 8; ++r) {
            const int qr = q0 + r + half * 8;
            const int c  = hh * 64 + dt * 16 + col16;
            Zw[((size_t)bi * L_ + qr) * D_ + c] = f2bf(acc[dt][r] * inv[r]);
        }
}

// ---------------------------------------------------------------------------
// Kernel 4: out = Z @ (Wo*mask), f32 output.
// ---------------------------------------------------------------------------
__global__ __launch_bounds__(256) void out_proj(
    const bf16_t* __restrict__ Zw, const bf16_t* __restrict__ WoT,
    float* __restrict__ out)
{
    const int lane  = threadIdx.x & 31;
    const int wave  = threadIdx.x >> 5;
    const int wid   = blockIdx.x * 8 + wave;
    const int ngrp  = wid & 7;
    const int mtile = wid >> 3;
    const int tok0  = mtile * 16;
    const int col16 = lane & 15;
    const int half  = lane >> 4;

    v8f acc[4];
#pragma unroll
    for (int nt = 0; nt < 4; ++nt) acc[nt] = vzero8();

    const bf16_t* arow = Zw + (size_t)(tok0 + col16) * D_;

    for (int kc = 0; kc < D_ / 32; ++kc) {
        v8bf lo = *(const v8bf*)(arow + kc * 32 + half * 16);
        v8bf hi = *(const v8bf*)(arow + kc * 32 + half * 16 + 8);
        v16bf a = cat16(lo, hi);
#pragma unroll
        for (int nt = 0; nt < 4; ++nt) {
            const bf16_t* bp = WoT + (size_t)(ngrp * 64 + nt * 16 + col16) * D_
                                   + kc * 32 + half * 16;
            v16bf bm = *(const v16bf*)bp;
            acc[nt] = wmma_bf16(a, bm, acc[nt]);
        }
    }
#pragma unroll
    for (int nt = 0; nt < 4; ++nt)
#pragma unroll
        for (int r = 0; r < 8; ++r) {
            const int tok = tok0 + r + half * 8;
            const int c = ngrp * 64 + nt * 16 + col16;
            out[(size_t)tok * D_ + c] = acc[nt][r];
        }
}

// ---------------------------------------------------------------------------
extern "C" void kernel_launch(void* const* d_in, const int* in_sizes, int n_in,
                              void* d_out, int out_size, void* d_ws, size_t ws_size,
                              hipStream_t stream)
{
    const float* x    = (const float*)d_in[0];
    const float* y    = (const float*)d_in[1];
    const float* Wq   = (const float*)d_in[2];
    const float* Wk   = (const float*)d_in[3];
    const float* Wv   = (const float*)d_in[4];
    const float* Wo   = (const float*)d_in[5];
    const float* mask = (const float*)d_in[6];
    float* out = (float*)d_out;

    const size_t WT_ELEMS  = (size_t)D_ * D_;
    const size_t XY_ELEMS  = (size_t)B_ * L_ * D_;
    const size_t QKV_ELEMS = (size_t)B_ * H_ * L_ * DKK;

    bf16_t* wt  = (bf16_t*)d_ws;
    bf16_t* xb  = wt  + 4 * WT_ELEMS;
    bf16_t* yb  = xb  + XY_ELEMS;
    bf16_t* qkv = yb  + XY_ELEMS;
    bf16_t* Zw  = qkv + 3 * QKV_ELEMS;

    // 0) x,y -> bf16
    cvt_xy<<<dim3((unsigned)(XY_ELEMS / 8 / 256), 2), 256, 0, stream>>>(x, y, xb, yb);

    // 1) weights: mask-fold + transpose + bf16 cast
    prep_weights<<<dim3((D_ * D_) / 256, 4), 256, 0, stream>>>(Wq, Wk, Wv, Wo, mask, wt);

    // 2) Q/K/V projections
    proj_qkv<<<dim3(512, 3), 256, 0, stream>>>(xb, yb, wt, qkv);

    // 3) flash attention
    flash_attn<<<dim3(B_ * H_ * (L_ / 128)), 256, 0, stream>>>(
        qkv, qkv + QKV_ELEMS, qkv + 2 * QKV_ELEMS, Zw);

    // 4) output projection
    out_proj<<<dim3(512), 256, 0, stream>>>(Zw, wt + 3 * WT_ELEMS, out);
}
